// LSTM_33818572488965
// MI455X (gfx1250) — matrix operand order repaired and checked by reading, not compiled, not run on hardware
//
#include <hip/hip_runtime.h>
#include <hip/hip_bf16.h>

// ---------------------------------------------------------------------------
// Types for CDNA5 WMMA bf16 path
// ---------------------------------------------------------------------------
typedef __bf16 bf16_t;
typedef bf16_t v16bf __attribute__((ext_vector_type(16)));
typedef bf16_t v8bf  __attribute__((ext_vector_type(8)));
typedef float  v8f   __attribute__((ext_vector_type(8)));

#define WMMA_BF16(a, b, c) \
  __builtin_amdgcn_wmma_f32_16x16x32_bf16(false, (a), false, (b), (short)0, (c), false, false)

// Problem constants
constexpr int Bb = 64;          // batch
constexpr int Tt = 512;         // time
constexpr int Ee = 512;         // embed
constexpr int Hh = 512;         // hidden
constexpr int NG = 4 * Hh;      // 2048 gate columns (i,f,g,o)
constexpr int MROWS = Bb * Tt;  // 32768 GEMM rows
constexpr int HCHUNK = 32;      // h columns per scan workgroup
constexpr int NWG_DIR = 16;     // scan workgroups per direction (16*32 = 512 = H)
constexpr int LDP = 40;         // padded LDS panel row stride (bf16): 80B = 20 banks

// ---------------------------------------------------------------------------
// WMMA fragment loaders (per documented gfx1250 VGPR layouts)
// A (16x32 bf16): lane<16 row m=lane holds K {0..7, 16..23}; lane>=16 holds {8..15, 24..31}
// B (32x16 bf16): lane n=lane&15 holds K {0..7, 16..23} (lane<16) or {8..15, 24..31} (lane>=16)
// C/D (16x16 f32): v8f, element r -> row = r + (lane>=16 ? 8 : 0), col = lane&15
// All loads are 16-byte units (safe for both global b128 and ds b128 paths).
// ---------------------------------------------------------------------------
__device__ __forceinline__ v16bf load_a_frag(const bf16_t* A, int lda, int lane) {
  const int m = lane & 15, half = lane >> 4;
  const bf16_t* p = A + (size_t)m * lda + half * 8;
  v8bf lo = *(const v8bf*)p;          // K = k0 + half*8 + 0..7
  v8bf hi = *(const v8bf*)(p + 16);   // K = k0 + 16 + half*8 + 0..7
  v16bf r;
#pragma unroll
  for (int i = 0; i < 8; ++i) { r[i] = lo[i]; r[i + 8] = hi[i]; }
  return r;
}

__device__ __forceinline__ v16bf load_b_frag(const bf16_t* W, int ldw, int lane) {
  // W is row-major [N, K]; fragment column n reads K {half*8+0..7, 16+half*8+0..7}.
  const int n = lane & 15, half = lane >> 4;
  const bf16_t* p = W + (size_t)n * ldw + half * 8;
  v8bf lo = *(const v8bf*)p;
  v8bf hi = *(const v8bf*)(p + 16);
  v16bf r;
#pragma unroll
  for (int i = 0; i < 8; ++i) { r[i] = lo[i]; r[i + 8] = hi[i]; }
  return r;
}

__device__ __forceinline__ float sigm(float x) { return 1.0f / (1.0f + __expf(-x)); }

// ---------------------------------------------------------------------------
// Elementwise helpers
// ---------------------------------------------------------------------------
__global__ void cvt_f32_bf16_kernel(const float* __restrict__ src,
                                    bf16_t* __restrict__ dst, size_t n) {
  size_t stride = (size_t)gridDim.x * blockDim.x;
  for (size_t i = (size_t)blockIdx.x * blockDim.x + threadIdx.x; i < n; i += stride)
    dst[i] = (bf16_t)src[i];
}

__global__ void bias_sum_kernel(const float* __restrict__ a, const float* __restrict__ b,
                                float* __restrict__ out, int n) {
  int i = blockIdx.x * blockDim.x + threadIdx.x;
  if (i < n) out[i] = a[i] + b[i];
}

// ---------------------------------------------------------------------------
// WMMA GEMM: C[M,N] = A[M,K](bf16) * W[N,K]^T(bf16) + bias[N]
// WG: 256 threads = 8 waves in 2x4 layout -> 128x128 WG tile, 64x32/wave.
// Per 32-wide k-chunk the 128x32 A and B panels are staged once through LDS
// (padded stride, 8KB each) and the next chunk's global loads are issued
// before the WMMA burst so VMEM latency overlaps matrix math.
// ---------------------------------------------------------------------------
__global__ __launch_bounds__(256, 1)
void wmma_gemm_bias_kernel(const bf16_t* __restrict__ A, const bf16_t* __restrict__ W,
                           const float* __restrict__ bias, float* __restrict__ C,
                           int M, int N, int K) {
  __shared__ bf16_t aPanel[128 * LDP];   // 10240 B
  __shared__ bf16_t bPanel[128 * LDP];   // 10240 B

  const int tid = threadIdx.x;
  const int lane = tid & 31;
  const int wave = tid >> 5;
  const int wm = wave >> 2;                 // 0..1
  const int wn = wave & 3;                  // 0..3
  const int tiles_n = N / 128;
  const int bm = blockIdx.x / tiles_n;
  const int bn = blockIdx.x % tiles_n;
  const int m_wg = bm * 128;
  const int n_wg = bn * 128;

  // Staging: 512 uint4 (8 bf16) items per panel -> 2 per thread per panel.
  const int ar0 = tid >> 2,          aq0 = tid & 3;
  const int ar1 = (tid + 256) >> 2,  aq1 = (tid + 256) & 3;
  const bf16_t* Abase = A + (size_t)m_wg * K;
  const bf16_t* Wbase = W + (size_t)n_wg * K;

  uint4 ra0, ra1, rb0, rb1;
  auto glb_load = [&](int k0) {
    ra0 = *(const uint4*)(Abase + (size_t)ar0 * K + k0 + aq0 * 8);
    ra1 = *(const uint4*)(Abase + (size_t)ar1 * K + k0 + aq1 * 8);
    rb0 = *(const uint4*)(Wbase + (size_t)ar0 * K + k0 + aq0 * 8);
    rb1 = *(const uint4*)(Wbase + (size_t)ar1 * K + k0 + aq1 * 8);
  };

  glb_load(0);
  v8f acc[4][2] = {};

  const int nk = K / 32;
  for (int kc = 0; kc < nk; ++kc) {
    __syncthreads();   // previous chunk's fragment reads complete
    *(uint4*)(aPanel + ar0 * LDP + aq0 * 8) = ra0;
    *(uint4*)(aPanel + ar1 * LDP + aq1 * 8) = ra1;
    *(uint4*)(bPanel + ar0 * LDP + aq0 * 8) = rb0;
    *(uint4*)(bPanel + ar1 * LDP + aq1 * 8) = rb1;
    __syncthreads();
    if (kc + 1 < nk) glb_load((kc + 1) * 32);   // overlap next loads with WMMA

    v16bf bf0 = load_b_frag(bPanel + (size_t)(wn * 32)      * LDP, LDP, lane);
    v16bf bf1 = load_b_frag(bPanel + (size_t)(wn * 32 + 16) * LDP, LDP, lane);
#pragma unroll
    for (int mt = 0; mt < 4; ++mt) {
      v16bf af = load_a_frag(aPanel + (size_t)(wm * 64 + mt * 16) * LDP, LDP, lane);
      acc[mt][0] = WMMA_BF16(af, bf0, acc[mt][0]);
      acc[mt][1] = WMMA_BF16(af, bf1, acc[mt][1]);
    }
  }

  const int m_base = m_wg + wm * 64;
  const int n_base = n_wg + wn * 32;
  const int n = lane & 15, mo = (lane >> 4) * 8;
#pragma unroll
  for (int nt = 0; nt < 2; ++nt) {
    const float bv = bias[n_base + nt * 16 + n];
#pragma unroll
    for (int mt = 0; mt < 4; ++mt)
#pragma unroll
      for (int r = 0; r < 8; ++r) {
        const int m = m_base + mt * 16 + mo + r;
        C[(size_t)m * N + n_base + nt * 16 + n] = acc[mt][nt][r] + bv;
      }
  }
}

// ---------------------------------------------------------------------------
// Persistent bidirectional LSTM scan (one layer, both directions).
// grid = 32 WGs: dir = blockIdx.x/16, h-chunk = blockIdx.x%16 (32 h columns).
// 8 waves/WG; wave w handles gate g=w/2, 16 gate columns. W_hh strip (16 rows
// x K=512 bf16) stays register-resident across all 512 timesteps.
// Per step: stage h (64x512 bf16) to LDS, 64 wmma/wave, i/f/g/o combine via
// LDS, write h chunk to global, monotonic-counter grid barrier per direction.
// Dynamic LDS: 64KB h + 32KB gates + 8KB cell state = 106496 B.
// ---------------------------------------------------------------------------
constexpr size_t SCAN_LDS_BYTES = (size_t)Bb * Hh * 2 + (size_t)Bb * 128 * 4 + (size_t)Bb * HCHUNK * 4;

__global__ __launch_bounds__(256, 1)
void lstm_scan_kernel(const bf16_t* __restrict__ Whh,   // [2][NG][Hh] bf16
                      const float* __restrict__ xg,     // [2][B*T][NG] f32 (bias included)
                      bf16_t* __restrict__ hbuf,        // [2][B][Hh] bf16 (zero-init)
                      bf16_t* __restrict__ yout,        // [B][T][2H] bf16 or nullptr
                      float* __restrict__ hfin,         // [2][B][Hh] f32 or nullptr
                      unsigned* __restrict__ barriers)  // [2] monotonic counters (zero-init)
{
  extern __shared__ char smem_raw[];
  bf16_t* h_lds = (bf16_t*)smem_raw;                                   // 64 KB
  float*  gbuf  = (float*)(smem_raw + (size_t)Bb * Hh * 2);            // 32 KB  [B][128]
  float*  c_lds = (float*)(smem_raw + (size_t)Bb * Hh * 2 + (size_t)Bb * 128 * 4); // 8 KB

  const int tid = threadIdx.x;
  const int lane = tid & 31;
  const int wave = tid >> 5;                  // 0..7
  const int dir = blockIdx.x / NWG_DIR;
  const int chunk = blockIdx.x % NWG_DIR;
  const int h0 = chunk * HCHUNK;
  const int g = wave >> 1, sub = wave & 1;
  const int row_base = g * Hh + h0 + sub * 16;       // gate-column base in [0,2048)

  // Register-resident W_hh strip: 16 rows x 512 K -> 16 bf16 fragments (128 VGPRs).
  const bf16_t* Wd = Whh + (size_t)dir * NG * Hh;
  v16bf bfr[16];
#pragma unroll
  for (int kc = 0; kc < 16; ++kc)
    bfr[kc] = load_b_frag(Wd + (size_t)row_base * Hh + kc * 32, Hh, lane);

  // zero cell state
  for (int i = tid; i < Bb * HCHUNK; i += 256) c_lds[i] = 0.0f;
  __syncthreads();

  const float* xgp = xg + (size_t)dir * MROWS * NG;
  bf16_t* hglob = hbuf + (size_t)dir * Bb * Hh;
  unsigned* barc = barriers + dir;
  const int cn = lane & 15, mo = (lane >> 4) * 8;
  const int colg = row_base + cn;          // global gate column
  const int col128 = wave * 16 + cn;       // column inside [i|f|g|o] x 32 buffer

  for (int it = 0; it < Tt; ++it) {
    const int t = (dir == 0) ? it : (Tt - 1 - it);

    // Stage h_prev (64x512 bf16 = 4096 uint4) into LDS cooperatively.
    {
      const uint4* hs = (const uint4*)hglob;
      uint4* hd = (uint4*)h_lds;
#pragma unroll
      for (int i = 0; i < 16; ++i) hd[tid + i * 256] = hs[tid + i * 256];
    }
    __syncthreads();

    // GEMM slice: [64 x 16] = h[64x512] @ Whh_strip^T
    v8f acc[4] = {};
#pragma unroll
    for (int kc = 0; kc < 16; ++kc) {
#pragma unroll
      for (int mt = 0; mt < 4; ++mt) {
        v16bf afr = load_a_frag(h_lds + (size_t)(mt * 16) * Hh + kc * 32, Hh, lane);
        acc[mt] = WMMA_BF16(afr, bfr[kc], acc[mt]);
      }
    }

    // Gate pre-activations (+ precomputed xg) into LDS for cross-gate combine.
#pragma unroll
    for (int mt = 0; mt < 4; ++mt)
#pragma unroll
      for (int r = 0; r < 8; ++r) {
        const int b = mt * 16 + mo + r;
        gbuf[b * 128 + col128] = acc[mt][r] + xgp[((size_t)b * Tt + t) * NG + colg];
      }
    __syncthreads();

    // Elementwise LSTM cell update for 64 x 32 state values.
#pragma unroll
    for (int i = 0; i < 8; ++i) {
      const int idx = tid + i * 256;     // 0..2047
      const int b = idx >> 5, j = idx & 31;
      const float iv = gbuf[b * 128 + j];
      const float fv = gbuf[b * 128 + 32 + j];
      const float gv = gbuf[b * 128 + 64 + j];
      const float ov = gbuf[b * 128 + 96 + j];
      const float c = sigm(fv) * c_lds[idx] + sigm(iv) * tanhf(gv);
      c_lds[idx] = c;
      const float h = sigm(ov) * tanhf(c);
      hglob[(size_t)b * Hh + h0 + j] = (bf16_t)h;
      if (yout) yout[((size_t)b * Tt + t) * (2 * Hh) + dir * Hh + h0 + j] = (bf16_t)h;
      if (hfin) hfin[(size_t)dir * Bb * Hh + (size_t)b * Hh + h0 + j] = h;
    }
    __threadfence();
    __syncthreads();

    // Grid barrier across the 16 WGs of this direction (monotonic counter).
    if (tid == 0) {
      atomicAdd(barc, 1u);
      const unsigned target = (unsigned)(it + 1) * NWG_DIR;
      while (atomicAdd(barc, 0u) < target) __builtin_amdgcn_s_sleep(2);
    }
    __syncthreads();
    __threadfence();
  }
}

// ---------------------------------------------------------------------------
// Final head: out[b] = concat(hT_f, hT_b) . lin_w + lin_b
// ---------------------------------------------------------------------------
__global__ __launch_bounds__(256)
void final_linear_kernel(const float* __restrict__ hfin,  // [2][B][Hh]
                         const float* __restrict__ lin_w, // [2H]
                         const float* __restrict__ lin_b, // [1]
                         float* __restrict__ out) {       // [B]
  __shared__ float red[256];
  const int b = blockIdx.x, tid = threadIdx.x;
  float s = 0.0f;
  for (int j = tid; j < 2 * Hh; j += 256) {
    const int d = j >> 9, jj = j & (Hh - 1);
    s += hfin[(size_t)d * Bb * Hh + (size_t)b * Hh + jj] * lin_w[j];
  }
  red[tid] = s;
  __syncthreads();
  for (int st = 128; st > 0; st >>= 1) {
    if (tid < st) red[tid] += red[tid + st];
    __syncthreads();
  }
  if (tid == 0) out[b] = red[0] + lin_b[0];
}

// ---------------------------------------------------------------------------
// Workspace layout (bytes, 256-aligned regions)
// ---------------------------------------------------------------------------
constexpr size_t aln(size_t x) { return (x + 255) & ~(size_t)255; }
constexpr size_t OFF_TEXTBF = 0;                                             // B*T*E bf16
constexpr size_t OFF_WIH0   = aln(OFF_TEXTBF + (size_t)MROWS * Ee * 2);      // 2*NG*E bf16
constexpr size_t OFF_WHH0   = aln(OFF_WIH0 + (size_t)2 * NG * Ee * 2);       // 2*NG*H bf16
constexpr size_t OFF_WIH1   = aln(OFF_WHH0 + (size_t)2 * NG * Hh * 2);       // 2*NG*2H bf16
constexpr size_t OFF_WHH1   = aln(OFF_WIH1 + (size_t)2 * NG * 2 * Hh * 2);   // 2*NG*H bf16
constexpr size_t OFF_B0     = aln(OFF_WHH1 + (size_t)2 * NG * Hh * 2);       // 2*NG f32
constexpr size_t OFF_B1     = aln(OFF_B0 + (size_t)2 * NG * 4);
constexpr size_t OFF_HBUF   = aln(OFF_B1 + (size_t)2 * NG * 4);              // 2*B*H bf16
constexpr size_t OFF_HFIN   = aln(OFF_HBUF + (size_t)2 * Bb * Hh * 2);       // 2*B*H f32
constexpr size_t OFF_BAR    = aln(OFF_HFIN + (size_t)2 * Bb * Hh * 4);       // barriers
constexpr size_t OFF_Y0     = aln(OFF_BAR + 256);                            // B*T*2H bf16
constexpr size_t OFF_XG     = aln(OFF_Y0 + (size_t)MROWS * 2 * Hh * 2);      // 2*M*NG f32

extern "C" void kernel_launch(void* const* d_in, const int* in_sizes, int n_in,
                              void* d_out, int out_size, void* d_ws, size_t ws_size,
                              hipStream_t stream) {
  (void)in_sizes; (void)n_in; (void)out_size; (void)ws_size;
  const float* text  = (const float*)d_in[0];
  const float* w_ih0 = (const float*)d_in[1];
  const float* w_hh0 = (const float*)d_in[2];
  const float* b_ih0 = (const float*)d_in[3];
  const float* b_hh0 = (const float*)d_in[4];
  const float* w_ih1 = (const float*)d_in[5];
  const float* w_hh1 = (const float*)d_in[6];
  const float* b_ih1 = (const float*)d_in[7];
  const float* b_hh1 = (const float*)d_in[8];
  const float* lin_w = (const float*)d_in[9];
  const float* lin_b = (const float*)d_in[10];

  char* ws = (char*)d_ws;
  bf16_t*   text_bf = (bf16_t*)(ws + OFF_TEXTBF);
  bf16_t*   wih0_bf = (bf16_t*)(ws + OFF_WIH0);
  bf16_t*   whh0_bf = (bf16_t*)(ws + OFF_WHH0);
  bf16_t*   wih1_bf = (bf16_t*)(ws + OFF_WIH1);
  bf16_t*   whh1_bf = (bf16_t*)(ws + OFF_WHH1);
  float*    b0      = (float*)(ws + OFF_B0);
  float*    b1      = (float*)(ws + OFF_B1);
  bf16_t*   hbuf    = (bf16_t*)(ws + OFF_HBUF);
  float*    hfin    = (float*)(ws + OFF_HFIN);
  unsigned* bar     = (unsigned*)(ws + OFF_BAR);
  bf16_t*   y0_bf   = (bf16_t*)(ws + OFF_Y0);
  float*    xg      = (float*)(ws + OFF_XG);

  // --- precision conversion / bias folding -------------------------------
  cvt_f32_bf16_kernel<<<2048, 256, 0, stream>>>(text,  text_bf, (size_t)MROWS * Ee);
  cvt_f32_bf16_kernel<<<1024, 256, 0, stream>>>(w_ih0, wih0_bf, (size_t)2 * NG * Ee);
  cvt_f32_bf16_kernel<<<1024, 256, 0, stream>>>(w_hh0, whh0_bf, (size_t)2 * NG * Hh);
  cvt_f32_bf16_kernel<<<1024, 256, 0, stream>>>(w_ih1, wih1_bf, (size_t)2 * NG * 2 * Hh);
  cvt_f32_bf16_kernel<<<1024, 256, 0, stream>>>(w_hh1, whh1_bf, (size_t)2 * NG * Hh);
  bias_sum_kernel<<<(2 * NG + 255) / 256, 256, 0, stream>>>(b_ih0, b_hh0, b0, 2 * NG);
  bias_sum_kernel<<<(2 * NG + 255) / 256, 256, 0, stream>>>(b_ih1, b_hh1, b1, 2 * NG);

  const int gemm_grid = (MROWS / 128) * (NG / 128);   // 256 * 16 = 4096 WGs

  // --- layer 0 -----------------------------------------------------------
  hipMemsetAsync(hbuf, 0, (size_t)2 * Bb * Hh * 2, stream);
  hipMemsetAsync(bar, 0, 256, stream);
  for (int dir = 0; dir < 2; ++dir)
    wmma_gemm_bias_kernel<<<gemm_grid, 256, 0, stream>>>(
        text_bf, wih0_bf + (size_t)dir * NG * Ee, b0 + (size_t)dir * NG,
        xg + (size_t)dir * MROWS * NG, MROWS, NG, Ee);
  lstm_scan_kernel<<<2 * NWG_DIR, 256, SCAN_LDS_BYTES, stream>>>(
      whh0_bf, xg, hbuf, y0_bf, nullptr, bar);

  // --- layer 1 -----------------------------------------------------------
  hipMemsetAsync(hbuf, 0, (size_t)2 * Bb * Hh * 2, stream);
  hipMemsetAsync(bar, 0, 256, stream);
  for (int dir = 0; dir < 2; ++dir)
    wmma_gemm_bias_kernel<<<gemm_grid, 256, 0, stream>>>(
        y0_bf, wih1_bf + (size_t)dir * NG * 2 * Hh, b1 + (size_t)dir * NG,
        xg + (size_t)dir * MROWS * NG, MROWS, NG, 2 * Hh);
  lstm_scan_kernel<<<2 * NWG_DIR, 256, SCAN_LDS_BYTES, stream>>>(
      whh1_bf, xg, hbuf, nullptr, hfin, bar);

  // --- head --------------------------------------------------------------
  final_linear_kernel<<<Bb, 256, 0, stream>>>(hfin, lin_w, lin_b, (float*)d_out);
}
